// B12xPagedAttention_88673894793844
// MI455X (gfx1250) — matrix operand order repaired and checked by reading, not compiled, not run on hardware
//
#include <hip/hip_runtime.h>

// ---- problem constants (shapes fixed by setup_inputs) ----
#define T_TOK   2048
#define B_SZ    4
#define QL      512
#define HID     4096
#define HQ      32
#define HKV     8
#define GQA     4
#define DHEAD   128
#define RD_     64
#define NQKV    6144   // (HQ + 2*HKV) * DHEAD
#define KVLEN   1024   // MP * PS
#define PS      16
#define MP      64

typedef __attribute__((ext_vector_type(16))) __bf16 bf16x16;
typedef __attribute__((ext_vector_type(8)))  __bf16 bf16x8;
typedef __attribute__((ext_vector_type(8)))  float  f32x8;

__device__ __forceinline__ bf16x16 join8(bf16x8 lo, bf16x8 hi) {
  return __builtin_shufflevector(lo, hi, 0, 1, 2, 3, 4, 5, 6, 7,
                                 8, 9, 10, 11, 12, 13, 14, 15);
}

// reductions over the 16 lanes of one wave half (rows of a WMMA C tile)
__device__ __forceinline__ float rmax16(float v) {
  v = fmaxf(v, __shfl_xor(v, 1, 32));
  v = fmaxf(v, __shfl_xor(v, 2, 32));
  v = fmaxf(v, __shfl_xor(v, 4, 32));
  v = fmaxf(v, __shfl_xor(v, 8, 32));
  return v;
}
__device__ __forceinline__ float rsum16(float v) {
  v += __shfl_xor(v, 1, 32);
  v += __shfl_xor(v, 2, 32);
  v += __shfl_xor(v, 4, 32);
  v += __shfl_xor(v, 8, 32);
  return v;
}

// ---------------------------------------------------------------------------
// One-time f32 -> bf16 conversion, 8 elements / thread (32B read, 16B write).
// ---------------------------------------------------------------------------
__global__ __launch_bounds__(256) void f32_to_bf16(const float* __restrict__ src,
                                                   __bf16* __restrict__ dst, int n8) {
  int i = blockIdx.x * 256 + threadIdx.x;
  if (i >= n8) return;
  const float4* s = (const float4*)src + (size_t)i * 2;
  float4 x0 = s[0], x1 = s[1];
  bf16x8 o;
  o[0] = (__bf16)x0.x; o[1] = (__bf16)x0.y; o[2] = (__bf16)x0.z; o[3] = (__bf16)x0.w;
  o[4] = (__bf16)x1.x; o[5] = (__bf16)x1.y; o[6] = (__bf16)x1.z; o[7] = (__bf16)x1.w;
  *((bf16x8*)dst + i) = o;
}

// ---------------------------------------------------------------------------
// C[M][N] = A[M][K] @ W[N][K]^T  (bf16 in, f32 out via WMMA f32 accumulate)
// 256 threads = 8 waves arranged 2(m) x 4(n); each wave owns a 16x16 tile.
// ---------------------------------------------------------------------------
__global__ __launch_bounds__(256) void gemm_bf16(const __bf16* __restrict__ A,
                                                 const __bf16* __restrict__ W,
                                                 float* __restrict__ C,
                                                 int M, int N, int K) {
  const int lane = threadIdx.x & 31;
  const int wave = threadIdx.x >> 5;
  const int hi = lane >> 4, ln = lane & 15;
  const int m0 = blockIdx.y * 32 + (wave >> 2) * 16;
  const int n0 = blockIdx.x * 64 + (wave & 3) * 16;
  const __bf16* arow = A + (size_t)(m0 + ln) * K;
  const __bf16* wrow = W + (size_t)(n0 + ln) * K;
  f32x8 acc = {};
  for (int kk = 0; kk < K; kk += 32) {
    __builtin_prefetch(arow + kk + 256, 0, 0);   // global_prefetch_b8
    __builtin_prefetch(wrow + kk + 256, 0, 0);
    // A 16x32 layout: lane row m0+ln; e -> K=kk+16*hi+e (contiguous)
    bf16x8 a0 = *(const bf16x8*)(arow + kk + 16 * hi);
    bf16x8 a1 = *(const bf16x8*)(arow + kk + 16 * hi + 8);
    // B 32x16 layout: lane col n0+ln; e -> K=kk+16*hi+e (contiguous 32B)
    bf16x16 b = *(const bf16x16*)(wrow + kk + 16 * hi);
    acc = __builtin_amdgcn_wmma_f32_16x16x32_bf16(false, join8(a0, a1), false, b,
                                                  (short)0, acc, false, false);
  }
#pragma unroll
  for (int r = 0; r < 8; ++r)
    C[(size_t)(m0 + r + 8 * hi) * N + n0 + ln] = acc[r];
}

// ---------------------------------------------------------------------------
// Per (token, head) RMSNorm + partial RoPE -> bf16, directly into the
// attention-friendly layouts.  hidx<HQ: Q head; <HQ+HKV: K head; else V head.
//   q_bf  [T][HQ][D]                 (A-operand rows for QK^T)
//   kall  [B][HKV][KVLEN][D]         (B-operand rows for QK^T)
//   vT    [B][HKV][D][KVLEN]         (B-operand rows for P*V, transposed)
// ---------------------------------------------------------------------------
__global__ __launch_bounds__(128) void rmsnorm_rope(const float* __restrict__ qkv,
                                                    const int* __restrict__ positions,
                                                    const float* __restrict__ cosb,
                                                    const float* __restrict__ sinb,
                                                    const float* __restrict__ qw,
                                                    const float* __restrict__ kw,
                                                    const int* __restrict__ cache_seqlens,
                                                    __bf16* __restrict__ q_bf,
                                                    __bf16* __restrict__ kall,
                                                    __bf16* __restrict__ vT) {
  const int t = blockIdx.y;
  const int hidx = blockIdx.x;           // 0..HQ+2*HKV-1
  const int d = threadIdx.x;             // 0..127
  const int b = t / QL, qi = t % QL;

  if (hidx >= HQ + HKV) {                // V head: plain copy, transposed scatter
    const int h = hidx - (HQ + HKV);
    const int slot = cache_seqlens[b] + qi;
    float v = qkv[(size_t)t * NQKV + (HQ + HKV) * DHEAD + h * DHEAD + d];
    vT[((size_t)(b * HKV + h) * DHEAD + d) * KVLEN + slot] = (__bf16)v;
    return;
  }

  const bool isq = hidx < HQ;
  const int h = isq ? hidx : hidx - HQ;
  const float* x = qkv + (size_t)t * NQKV + (isq ? h * DHEAD : HQ * DHEAD + h * DHEAD);
  const float* wgt = isq ? qw : kw;
  float v = x[d];
  float ss = v * v;
#pragma unroll
  for (int m = 1; m < 32; m <<= 1) ss += __shfl_xor(ss, m, 32);
  __shared__ float red[4];
  if ((threadIdx.x & 31) == 0) red[threadIdx.x >> 5] = ss;
  __syncthreads();
  const float rn = rsqrtf((red[0] + red[1] + red[2] + red[3]) * (1.0f / DHEAD) + 1e-6f);
  const int pos = positions[t];
  float out;
  if (d < RD_ / 2) {
    float n1 = x[d] * rn * wgt[d];
    float n2 = x[d + RD_ / 2] * rn * wgt[d + RD_ / 2];
    out = n1 * cosb[pos * (RD_ / 2) + d] - n2 * sinb[pos * (RD_ / 2) + d];
  } else if (d < RD_) {
    int i = d - RD_ / 2;
    float n1 = x[i] * rn * wgt[i];
    float n2 = x[d] * rn * wgt[d];
    out = n2 * cosb[pos * (RD_ / 2) + i] + n1 * sinb[pos * (RD_ / 2) + i];
  } else {
    out = v * rn * wgt[d];
  }
  if (isq) {
    q_bf[(size_t)t * (HQ * DHEAD) + h * DHEAD + d] = (__bf16)out;
  } else {
    const int slot = cache_seqlens[b] + qi;
    kall[((size_t)(b * HKV + h) * KVLEN + slot) * DHEAD + d] = (__bf16)out;
  }
}

// ---------------------------------------------------------------------------
// Resolve the page table ONCE: cached K/V (kpos < seqlen) -> bf16 kall / vT.
// grid (key, hkv, b), 128 threads = one head-dim row.
// ---------------------------------------------------------------------------
__global__ __launch_bounds__(128) void gather_kv(const float* __restrict__ k_cache,
                                                 const float* __restrict__ v_cache,
                                                 const int* __restrict__ page_table,
                                                 const int* __restrict__ cache_seqlens,
                                                 __bf16* __restrict__ kall,
                                                 __bf16* __restrict__ vT) {
  const int key = blockIdx.x;
  const int hkv = blockIdx.y;
  const int b = blockIdx.z;
  const int d = threadIdx.x;
  const int seqlen = cache_seqlens[b];
  if (key >= seqlen) return;
  const int page = page_table[b * MP + (key >> 4)];
  const size_t src = ((size_t)((size_t)page * PS + (key & (PS - 1))) * HKV + hkv) * DHEAD + d;
  kall[((size_t)(b * HKV + hkv) * KVLEN + key) * DHEAD + d] = (__bf16)k_cache[src];
  vT[((size_t)(b * HKV + hkv) * DHEAD + d) * KVLEN + key] = (__bf16)v_cache[src];
}

// ---------------------------------------------------------------------------
// Causal GQA attention, flash-style online softmax, all-bf16 operands.
// One wave per (16 queries, q-head, batch).
// ---------------------------------------------------------------------------
__global__ __launch_bounds__(32) void paged_attn(const __bf16* __restrict__ q_bf,
                                                 const __bf16* __restrict__ kall,
                                                 const __bf16* __restrict__ vT,
                                                 const int* __restrict__ cache_seqlens,
                                                 __bf16* __restrict__ attn_bf) {
  const int lane = threadIdx.x;
  const int hi = lane >> 4, ln = lane & 15;
  const int b = blockIdx.z, h = blockIdx.y;
  const int q0 = blockIdx.x * 16;
  const int hkv = h / GQA;
  const int seqlen = cache_seqlens[b];
  const __bf16* kbase = kall + (size_t)(b * HKV + hkv) * KVLEN * DHEAD;
  const __bf16* vbase = vT + (size_t)(b * HKV + hkv) * DHEAD * KVLEN;

  // Resident Q A-fragments: 4 chunks of 32 over DHEAD.
  bf16x16 qa[4];
  {
    const __bf16* qrow = q_bf + (size_t)(b * QL + q0 + ln) * (HQ * DHEAD) + h * DHEAD;
#pragma unroll
    for (int dc = 0; dc < 4; ++dc) {
      bf16x8 a0 = *(const bf16x8*)(qrow + dc * 32 + 16 * hi);
      bf16x8 a1 = *(const bf16x8*)(qrow + dc * 32 + 16 * hi + 8);
      qa[dc] = join8(a0, a1);
    }
  }

  f32x8 acc[8] = {};                     // O tile: 16q x 128d
  float m_run[8], s_run[8];
#pragma unroll
  for (int r = 0; r < 8; ++r) { m_run[r] = -3.0e38f; s_run[r] = 0.0f; }

  __shared__ __bf16 Pld[16][32];
  const float scale = 0.08838834764831845f;   // 1/sqrt(128)
  const f32x8 zero = {};

  for (int key0 = 0; key0 < KVLEN; key0 += 32) {
    if (key0 > seqlen + q0 + 15) break;        // fully-masked tail (uniform)

    // ---- S = Q @ K^T, two 16-key subtiles ----
    f32x8 sacc[2] = {zero, zero};
#pragma unroll
    for (int st = 0; st < 2; ++st) {
      const __bf16* krow = kbase + (size_t)(key0 + st * 16 + ln) * DHEAD;
#pragma unroll
      for (int dc = 0; dc < 4; ++dc) {
        bf16x16 kb = *(const bf16x16*)(krow + dc * 32 + 16 * hi);
        sacc[st] = __builtin_amdgcn_wmma_f32_16x16x32_bf16(false, qa[dc], false, kb,
                                                           (short)0, sacc[st], false, false);
      }
    }

    // ---- causal mask + online softmax (rows live in 16-lane halves) ----
    float p[2][8];
#pragma unroll
    for (int r = 0; r < 8; ++r) {
      const int row = r + 8 * hi;
      const int pos = seqlen + q0 + row;
      float s0 = sacc[0][r] * scale; if (key0 + ln > pos)      s0 = -3.0e38f;
      float s1 = sacc[1][r] * scale; if (key0 + 16 + ln > pos) s1 = -3.0e38f;
      float mloc = rmax16(fmaxf(s0, s1));
      float mnew = fmaxf(m_run[r], mloc);
      float fac = __expf(m_run[r] - mnew);
      float p0 = __expf(s0 - mnew);
      float p1 = __expf(s1 - mnew);
      float rs = rsum16(p0 + p1);
      s_run[r] = s_run[r] * fac + rs;
      m_run[r] = mnew;
#pragma unroll
      for (int j = 0; j < 8; ++j) acc[j][r] *= fac;
      p[0][r] = p0; p[1][r] = p1;
    }

    // ---- C-layout P -> LDS (bf16) -> A-layout fragment ----
#pragma unroll
    for (int r = 0; r < 8; ++r) {
      Pld[r + 8 * hi][ln]      = (__bf16)p[0][r];
      Pld[r + 8 * hi][16 + ln] = (__bf16)p[1][r];
    }
    __syncthreads();
    bf16x8 p0v = *(const bf16x8*)&Pld[ln][16 * hi];
    bf16x8 p1v = *(const bf16x8*)&Pld[ln][16 * hi + 8];
    bf16x16 pa = join8(p0v, p1v);
    __syncthreads();

    // ---- O += P @ V  (vT rows give contiguous 16-key runs per d-column) ----
#pragma unroll
    for (int j = 0; j < 8; ++j) {
      bf16x16 vb = *(const bf16x16*)(vbase + (size_t)(j * 16 + ln) * KVLEN + key0 + 16 * hi);
      acc[j] = __builtin_amdgcn_wmma_f32_16x16x32_bf16(false, pa, false, vb,
                                                       (short)0, acc[j], false, false);
    }
  }

  // ---- normalize + store (bf16, feeds the O-projection GEMM) ----
#pragma unroll
  for (int r = 0; r < 8; ++r) {
    const int row = r + 8 * hi;
    const float inv = 1.0f / s_run[r];
#pragma unroll
    for (int j = 0; j < 8; ++j)
      attn_bf[(size_t)(b * QL + q0 + row) * (HQ * DHEAD) + h * DHEAD + j * 16 + ln] =
          (__bf16)(acc[j][r] * inv);
  }
}

// ---------------------------------------------------------------------------
extern "C" void kernel_launch(void* const* d_in, const int* in_sizes, int n_in,
                              void* d_out, int out_size, void* d_ws, size_t ws_size,
                              hipStream_t stream) {
  (void)in_sizes; (void)n_in; (void)out_size; (void)ws_size;
  const float* hidden        = (const float*)d_in[0];
  const float* cosb          = (const float*)d_in[1];
  const float* sinb          = (const float*)d_in[2];
  const int*   positions     = (const int*)d_in[3];
  const float* k_cache       = (const float*)d_in[4];
  const float* v_cache       = (const float*)d_in[5];
  const int*   page_table    = (const int*)d_in[6];
  const int*   cache_seqlens = (const int*)d_in[7];
  // d_in[8] = cu_seqlens_q (uniform QL; unused)
  const float* qkv_weight    = (const float*)d_in[9];
  const float* o_weight      = (const float*)d_in[10];
  const float* q_norm_w      = (const float*)d_in[11];
  const float* k_norm_w      = (const float*)d_in[12];

  // ---- workspace layout ----
  char* ws = (char*)d_ws;
  float* qkv_ws = (float*)ws;                    ws += (size_t)T_TOK * NQKV * 4;       // 50.3 MB
  __bf16* hid_bf  = (__bf16*)ws;                 ws += (size_t)T_TOK * HID * 2;        // 16.8 MB
  __bf16* wqkv_bf = (__bf16*)ws;                 ws += (size_t)NQKV * HID * 2;         // 50.3 MB
  __bf16* wo_bf   = (__bf16*)ws;                 ws += (size_t)HID * HID * 2;          // 33.6 MB
  __bf16* q_bf    = (__bf16*)ws;                 ws += (size_t)T_TOK * HQ * DHEAD * 2; // 16.8 MB
  __bf16* kall    = (__bf16*)ws;                 ws += (size_t)B_SZ * HKV * KVLEN * DHEAD * 2; // 8.4 MB
  __bf16* vT      = (__bf16*)ws;                 ws += (size_t)B_SZ * HKV * KVLEN * DHEAD * 2; // 8.4 MB
  __bf16* attn_bf = (__bf16*)ws;                 // 16.8 MB

  // 1) one-time f32 -> bf16 conversions
  {
    int n8;
    n8 = (T_TOK * HID) / 8;
    f32_to_bf16<<<(n8 + 255) / 256, 256, 0, stream>>>(hidden, hid_bf, n8);
    n8 = (NQKV * HID) / 8;
    f32_to_bf16<<<(n8 + 255) / 256, 256, 0, stream>>>(qkv_weight, wqkv_bf, n8);
    n8 = (HID * HID) / 8;
    f32_to_bf16<<<(n8 + 255) / 256, 256, 0, stream>>>(o_weight, wo_bf, n8);
  }
  // 2) QKV projection (f32 output for accurate RMS statistics)
  gemm_bf16<<<dim3(NQKV / 64, T_TOK / 32), 256, 0, stream>>>(
      hid_bf, wqkv_bf, qkv_ws, T_TOK, NQKV, HID);
  // 3) RMSNorm + RoPE -> bf16 attention layouts (new tokens)
  rmsnorm_rope<<<dim3(HQ + 2 * HKV, T_TOK), 128, 0, stream>>>(
      qkv_ws, positions, cosb, sinb, q_norm_w, k_norm_w, cache_seqlens,
      q_bf, kall, vT);
  // 4) paged cache -> bf16 attention layouts (cached tokens, page table once)
  gather_kv<<<dim3(QL, HKV, B_SZ), 128, 0, stream>>>(
      k_cache, v_cache, page_table, cache_seqlens, kall, vT);
  // 5) attention
  paged_attn<<<dim3(QL / 16, HQ, B_SZ), 32, 0, stream>>>(
      q_bf, kall, vT, cache_seqlens, attn_bf);
  // 6) O projection -> f32 output
  gemm_bf16<<<dim3(HID / 64, T_TOK / 32), 256, 0, stream>>>(
      attn_bf, wo_bf, (float*)d_out, T_TOK, HID, HID);
}
